// DifferentiableHistogram_11398843203812
// MI455X (gfx1250) — matrix operand order repaired and checked by reading, not compiled, not run on hardware
//
#include <hip/hip_runtime.h>
#include <hip/hip_bf16.h>

typedef __attribute__((ext_vector_type(16))) _Float16 v16h;
typedef __attribute__((ext_vector_type(8)))  _Float16 v8h;
typedef __attribute__((ext_vector_type(8)))  float    v8f;

#define HW      65536               // 256*256 pixels per channel plane
#define KBINS   512                 // 8^3 bins
#define INV7    (1.0f / 7.0f)       // grid spacing of linspace(0,1,8)
// exp(-d^2/(2*sigma^2)) = exp2(-d^2 * log2(e)/(2*sigma^2)), sigma=0.02
#define EXPC    1803.3687f

// ---------------------------------------------------------------------------
// Kernel 0: zero the accumulator (d_out is poisoned by the harness)
// ---------------------------------------------------------------------------
__global__ void hist_zero_kernel(float* __restrict__ out) {
    out[blockIdx.x * 256 + threadIdx.x] = 0.0f;
}

// ---------------------------------------------------------------------------
// Kernel 1: separable soft-histogram via WMMA rank-1-update GEMM.
//   A (16x32 f16): rows i (8 used) x 32 pixels  -> normalized wx (corr folded)
//   B (32x16 f16): 32 pixels x 16 cols (4 tiles) -> vy[j]*wz[k]
//   C (16x16 f32): partial histogram tile, atomically reduced at the end.
// One wave handles 32 pixels per iteration; block = 8 waves, each with a
// private LDS slab (wave-local s_wait_dscnt ordering only, no __syncthreads).
// sched_barriers keep all four B fragments live simultaneously so the four
// WMMAs issue back-to-back without A/B WAR hazard v_nops.
// ---------------------------------------------------------------------------
__global__ __launch_bounds__(256) void hist_wmma_kernel(
        const float* __restrict__ x, float* __restrict__ out)
{
    const int lane = threadIdx.x & 31;
    const int wave = threadIdx.x >> 5;
    const int b    = blockIdx.y;                 // batch index (0..3)
    const int blockStart = blockIdx.x * 2048;    // 32 blocks cover 65536 pixels
    const float* xb = x + b * (3 * HW);

    // per-wave slab: aT[16][32] + vyT[8][32] + wzT[8][32]  (f16) = 2 KB
    __shared__ _Float16 lds[8][16 * 32 + 8 * 32 + 8 * 32];
    _Float16* aT  = &lds[wave][0];
    _Float16* vyT = aT + 16 * 32;
    _Float16* wzT = vyT + 8 * 32;

    // zero padded A rows 8..15 once (M-dim padding of the 8-row histogram)
    #pragma unroll
    for (int r = 8; r < 16; ++r) aT[r * 32 + lane] = (_Float16)0.0f;

    v8f c0 = {}, c1 = {}, c2 = {}, c3 = {};

    const int m  = lane & 15;   // row (A) / column-in-tile (B)
    const int hi = lane >> 4;   // lane half selects K sub-range
    const int k  = m & 7;       // z-bin: tile-invariant
    const int j0 = m >> 3;      // y-bin base; tile t uses j = j0 + 2t

    for (int it = 0; it < 8; ++it) {
        const int n = blockStart + it * 256 + wave * 32 + lane;

        const float px = xb[n];
        const float py = xb[HW + n];
        const float pz = xb[2 * HW + n];
        if (it < 7) {
            __builtin_prefetch(xb + n + 256, 0, 0);  // global_prefetch_b8
        }

        // 24 transcendentals per pixel instead of 512
        float wx[8], wy[8], wz[8];
        float Sx = 0.0f, Sy = 0.0f, Sz = 0.0f;
        #pragma unroll
        for (int i = 0; i < 8; ++i) {
            const float ci = (float)i * INV7;
            float dx = px - ci; wx[i] = exp2f(-dx * dx * EXPC); Sx += wx[i];
            float dy = py - ci; wy[i] = exp2f(-dy * dy * EXPC); Sy += wy[i];
            float dz = pz - ci; wz[i] = exp2f(-dz * dz * EXPC); Sz += wz[i];
        }
        const float S  = Sx * Sy * Sz;
        const float ax = (S / (S + 1e-8f)) / Sx;  // epsilon correction folded in
        const float by = 1.0f / Sy;
        const float cz = 1.0f / Sz;

        // stage normalized per-axis weights transposed: [axis-bin][pixel]
        #pragma unroll
        for (int i = 0; i < 8; ++i) {
            aT [i * 32 + lane] = (_Float16)(wx[i] * ax);
            vyT[i * 32 + lane] = (_Float16)(wy[i] * by);
            wzT[i * 32 + lane] = (_Float16)(wz[i] * cz);
        }
        // wave-local producer->consumer ordering (CDNA5 split DS counter)
        asm volatile("s_wait_dscnt 0x0" ::: "memory");
        __builtin_amdgcn_wave_barrier();

        // A fragment (shared by all 4 N-tiles): contiguous ds_load_b128 pairs
        v8h a0 = *(const v8h*)(aT + m * 32 + hi * 16);       // K = hi*16 .. +7
        v8h a1 = *(const v8h*)(aT + m * 32 + hi * 16 + 8);   // K = hi*16+8 ..
        v16h A = __builtin_shufflevector(a0, a1,
                    0, 1, 2, 3, 4, 5, 6, 7, 8, 9, 10, 11, 12, 13, 14, 15);

        // z fragment is identical for all tiles (k does not depend on t)
        v8h z0 = *(const v8h*)(wzT + k * 32 + hi * 16);
        v8h z1 = *(const v8h*)(wzT + k * 32 + hi * 16 + 8);

        // build all four independent B fragments, then issue 4 WMMAs
        v16h B0, B1, B2, B3;
        {
            v8h y0, y1, p0, p1;
            y0 = *(const v8h*)(vyT + (j0 + 0) * 32 + hi * 16);
            y1 = *(const v8h*)(vyT + (j0 + 0) * 32 + hi * 16 + 8);
            p0 = y0 * z0; p1 = y1 * z1;          // v_pk_mul_f16
            B0 = __builtin_shufflevector(p0, p1,
                    0, 1, 2, 3, 4, 5, 6, 7, 8, 9, 10, 11, 12, 13, 14, 15);
            y0 = *(const v8h*)(vyT + (j0 + 2) * 32 + hi * 16);
            y1 = *(const v8h*)(vyT + (j0 + 2) * 32 + hi * 16 + 8);
            p0 = y0 * z0; p1 = y1 * z1;
            B1 = __builtin_shufflevector(p0, p1,
                    0, 1, 2, 3, 4, 5, 6, 7, 8, 9, 10, 11, 12, 13, 14, 15);
            y0 = *(const v8h*)(vyT + (j0 + 4) * 32 + hi * 16);
            y1 = *(const v8h*)(vyT + (j0 + 4) * 32 + hi * 16 + 8);
            p0 = y0 * z0; p1 = y1 * z1;
            B2 = __builtin_shufflevector(p0, p1,
                    0, 1, 2, 3, 4, 5, 6, 7, 8, 9, 10, 11, 12, 13, 14, 15);
            y0 = *(const v8h*)(vyT + (j0 + 6) * 32 + hi * 16);
            y1 = *(const v8h*)(vyT + (j0 + 6) * 32 + hi * 16 + 8);
            p0 = y0 * z0; p1 = y1 * z1;
            B3 = __builtin_shufflevector(p0, p1,
                    0, 1, 2, 3, 4, 5, 6, 7, 8, 9, 10, 11, 12, 13, 14, 15);
        }
        // fence: all four B fragments must be materialized (and live) here
        __builtin_amdgcn_sched_barrier(0);

        c0 = __builtin_amdgcn_wmma_f32_16x16x32_f16(false, A, false, B0,
                                                    (short)0, c0, false, false);
        c1 = __builtin_amdgcn_wmma_f32_16x16x32_f16(false, A, false, B1,
                                                    (short)0, c1, false, false);
        c2 = __builtin_amdgcn_wmma_f32_16x16x32_f16(false, A, false, B2,
                                                    (short)0, c2, false, false);
        c3 = __builtin_amdgcn_wmma_f32_16x16x32_f16(false, A, false, B3,
                                                    (short)0, c3, false, false);
        // fence: keep next-iteration VALU from sliding into the WMMA group
        __builtin_amdgcn_sched_barrier(0);
        __builtin_amdgcn_wave_barrier();  // keep next-iter stores behind loads
    }

    // C layout: VGPR r, lane l<16 -> hist[r*64 + t*16 + l]; rows 8..15 are zero
    if (lane < 16) {
        float* o = out + b * KBINS;
        #pragma unroll
        for (int r = 0; r < 8; ++r) {
            atomicAdd(o + r * 64 +  0 + m, c0[r]);
            atomicAdd(o + r * 64 + 16 + m, c1[r]);
            atomicAdd(o + r * 64 + 32 + m, c2[r]);
            atomicAdd(o + r * 64 + 48 + m, c3[r]);
        }
    }
}

// ---------------------------------------------------------------------------
// Kernel 2: per-batch in-place normalization: h /= (sum(h) + 1e-8)
// ---------------------------------------------------------------------------
__global__ __launch_bounds__(512) void hist_norm_kernel(float* __restrict__ out) {
    __shared__ float red[512];
    const int t = threadIdx.x;
    const int b = blockIdx.x;
    const float v = out[b * KBINS + t];
    red[t] = v;
    __syncthreads();
    #pragma unroll
    for (int s = 256; s > 0; s >>= 1) {
        if (t < s) red[t] += red[t + s];
        __syncthreads();
    }
    out[b * KBINS + t] = v / (red[0] + 1e-8f);
}

// ---------------------------------------------------------------------------
extern "C" void kernel_launch(void* const* d_in, const int* in_sizes, int n_in,
                              void* d_out, int out_size, void* d_ws, size_t ws_size,
                              hipStream_t stream) {
    (void)in_sizes; (void)n_in; (void)out_size; (void)d_ws; (void)ws_size;
    const float* x = (const float*)d_in[0];   // (4,3,256,256) fp32
    // d_in[1] (bin_centers) is a regular linspace grid -> regenerated analytically
    float* out = (float*)d_out;               // (4,512) fp32

    hist_zero_kernel<<<dim3(8), dim3(256), 0, stream>>>(out);
    hist_wmma_kernel<<<dim3(32, 4), dim3(256), 0, stream>>>(x, out);
    hist_norm_kernel<<<dim3(4), dim3(512), 0, stream>>>(out);
}
